// KnowledgeAttentionLayer_23476291240054
// MI455X (gfx1250) — matrix operand order repaired and checked by reading, not compiled, not run on hardware
//
#include <hip/hip_runtime.h>
#include <hip/hip_bf16.h>

typedef __bf16 bf16;
typedef __attribute__((ext_vector_type(16))) __bf16 v16bf;
typedef __attribute__((ext_vector_type(8)))  float   v8f;
typedef __attribute__((ext_vector_type(8)))  __bf16  bf16x8;

#define NEGV (-1e30f)

// ---------------------------------------------------------------------------
// Block reduction helpers (blockDim.x == 256)
// ---------------------------------------------------------------------------
__device__ __forceinline__ float blk_sum(float v, float* red) {
    red[threadIdx.x] = v; __syncthreads();
    for (int s = 128; s > 0; s >>= 1) {
        if (threadIdx.x < s) red[threadIdx.x] += red[threadIdx.x + s];
        __syncthreads();
    }
    float r = red[0]; __syncthreads();
    return r;
}
__device__ __forceinline__ float blk_max(float v, float* red) {
    red[threadIdx.x] = v; __syncthreads();
    for (int s = 128; s > 0; s >>= 1) {
        if (threadIdx.x < s) red[threadIdx.x] = fmaxf(red[threadIdx.x], red[threadIdx.x + s]);
        __syncthreads();
    }
    float r = red[0]; __syncthreads();
    return r;
}

// Combine two contiguous 8x bf16 LDS chunks into one 16-element WMMA operand.
__device__ __forceinline__ v16bf ld_frag(const bf16* p0, const bf16* p1) {
    bf16x8 lo = *(const bf16x8*)p0;
    bf16x8 hi = *(const bf16x8*)p1;
    v16bf r;
#pragma unroll
    for (int j = 0; j < 8; ++j) { r[j] = lo[j]; r[j + 8] = hi[j]; }
    return r;
}

// ---------------------------------------------------------------------------
// Batched/strided bf16 WMMA GEMM (B always given as [N,K] "transposed"):
//   C = alpha * A @ B^T (+ bias);  A: [M,K] (lda), B: [N,K] (ldb), C: [M,N] (ldc)
//   batch z decodes (b, h): ptr += b*s?b + h*s?h
//   256 threads = 8 waves; block tile 128x64; each wave computes 32x32 via
//   4 x v_wmma_f32_16x16x32_bf16 per K-step of 32.
//   LDS is laid out so every fragment read is a 16B ds_load_b128:
//     As[128][32] row-major (A lane runs are contiguous), BsT[64][32] N-major.
// ---------------------------------------------------------------------------
template <bool OUTBF>
__global__ __launch_bounds__(256) void gemm_wmma_bf16(
    const bf16* __restrict__ A, const bf16* __restrict__ Bm, void* __restrict__ Cp,
    int M, int N, int K, int lda, int ldb, int ldc,
    long long sAb, long long sAh, long long sBb, long long sBh,
    long long sCb, long long sCh, int H2, float alpha,
    const float* __restrict__ bias)
{
    __shared__ bf16 As[128][32];   // 8 KB
    __shared__ bf16 BsT[64][32];   // 4 KB (BsT[n][k] = B[n][k])

    const int z = blockIdx.z;
    const int b = z / H2;
    const int h = z - b * H2;
    A  += (long long)b * sAb + (long long)h * sAh;
    Bm += (long long)b * sBb + (long long)h * sBh;
    const long long coff = (long long)b * sCb + (long long)h * sCh;

    const int tm = blockIdx.y * 128;
    const int tn = blockIdx.x * 64;
    const int tid = threadIdx.x;
    const int lane = tid & 31;
    const int wid = tid >> 5;
    const int wr = wid >> 1;   // 0..3: 32-row strip
    const int wc = wid & 1;    // 0..1: 32-col strip

    v8f acc00 = {}, acc01 = {}, acc10 = {}, acc11 = {};

    // staging indices
    const int ar = tid >> 1;            // 0..127
    const int ac = (tid & 1) << 4;      // 0 or 16
    const int bn = tid >> 2;            // 0..63
    const int bk = (tid & 3) << 3;      // 0,8,16,24

    // fragment indices
    const int fm = lane & 15;
    const int ka = (lane >> 4) << 3;    // 0 or 8
    const int kb = (lane >> 4) << 4;    // 0 or 16

    for (int k0 = 0; k0 < K; k0 += 32) {
        {   // stage A tile 128x32: two b128 loads + two b128 LDS stores per thread
            const int gm = tm + ar;
            bf16x8 v0 = {}, v1 = {};
            if (gm < M) {
                const bf16* ap = A + (long long)gm * lda + (k0 + ac);
                v0 = *(const bf16x8*)(ap);
                v1 = *(const bf16x8*)(ap + 8);
                if (k0 + 32 < K) __builtin_prefetch(ap + 32, 0, 1);
            }
            *(bf16x8*)(&As[ar][ac])     = v0;
            *(bf16x8*)(&As[ar][ac + 8]) = v1;
        }
        {   // stage B tile [N=64][K=32]: one b128 load + one b128 LDS store
            const int gn = tn + bn;
            bf16x8 v = {};
            if (gn < N) v = *(const bf16x8*)(Bm + (long long)gn * ldb + (k0 + bk));
            *(bf16x8*)(&BsT[bn][bk]) = v;
        }
        __syncthreads();

        const int r0 = wr * 32 + fm;
        const int c0 = wc * 32 + fm;
        v16bf a0 = ld_frag(&As[r0][ka],       &As[r0][ka + 16]);
        v16bf a1 = ld_frag(&As[r0 + 16][ka],  &As[r0 + 16][ka + 16]);
        v16bf b0 = ld_frag(&BsT[c0][ka],      &BsT[c0][ka + 16]);
        v16bf b1 = ld_frag(&BsT[c0 + 16][ka], &BsT[c0 + 16][ka + 16]);

        acc00 = __builtin_amdgcn_wmma_f32_16x16x32_bf16(false, a0, false, b0,
                                                        (short)0, acc00, false, false);
        acc01 = __builtin_amdgcn_wmma_f32_16x16x32_bf16(false, a0, false, b1,
                                                        (short)0, acc01, false, false);
        acc10 = __builtin_amdgcn_wmma_f32_16x16x32_bf16(false, a1, false, b0,
                                                        (short)0, acc10, false, false);
        acc11 = __builtin_amdgcn_wmma_f32_16x16x32_bf16(false, a1, false, b1,
                                                        (short)0, acc11, false, false);
        __syncthreads();
    }

    // C/D layout: element e -> M = e + 8*(lane>=16), N = lane&15
    const int lrow8 = (lane >> 4) << 3;
    const int lncol = lane & 15;
#pragma unroll
    for (int mi = 0; mi < 2; ++mi) {
        const int rbase = tm + wr * 32 + mi * 16 + lrow8;
#pragma unroll
        for (int ni = 0; ni < 2; ++ni) {
            const int col = tn + wc * 32 + ni * 16 + lncol;
            const v8f acc = (mi == 0) ? (ni == 0 ? acc00 : acc01)
                                      : (ni == 0 ? acc10 : acc11);
            const float bia = (bias && col < N) ? bias[col] : 0.f;
            if (col >= N) continue;
#pragma unroll
            for (int e = 0; e < 8; ++e) {
                const int row = rbase + e;
                if (row >= M) continue;
                const float v = acc[e] * alpha + bia;
                const long long o = coff + (long long)row * ldc + col;
                if (OUTBF) ((bf16*)Cp)[o] = (bf16)v;
                else       ((float*)Cp)[o] = v;
            }
        }
    }
}

// ---------------------------------------------------------------------------
// Elementwise / reduction support kernels
// ---------------------------------------------------------------------------
__global__ void k_f2bf(const float* __restrict__ x, bf16* __restrict__ y, int n) {
    int i = blockIdx.x * 256 + threadIdx.x;
    if (i < n) y[i] = (bf16)x[i];
}

// Transposed convert: src f32 [Bt, R, C] -> dst bf16 [Bt, C, R]
__global__ void k_cvtT(const float* __restrict__ src, bf16* __restrict__ dst,
                       int Bt, int R, int C) {
    int i = blockIdx.x * 256 + threadIdx.x;
    if (i >= Bt * R * C) return;
    const int c = i % C;
    const int t = i / C;
    const int r = t % R;
    const int b = t / R;
    dst[((long long)b * C + c) * R + r] = (bf16)src[i];
}

// Tiled bf16 transpose: src [R, C] -> dst [C, R]
__global__ __launch_bounds__(256) void k_tr_bf(const bf16* __restrict__ src,
                                               bf16* __restrict__ dst, int R, int C) {
    __shared__ bf16 t[32][33];
    const int lx = threadIdx.x & 31;
    const int ly = threadIdx.x >> 5;       // 0..7
#pragma unroll
    for (int j = 0; j < 32; j += 8) {
        const int r = blockIdx.y * 32 + ly + j;
        const int c = blockIdx.x * 32 + lx;
        t[ly + j][lx] = (r < R && c < C) ? src[(long long)r * C + c] : (bf16)0.f;
    }
    __syncthreads();
#pragma unroll
    for (int j = 0; j < 32; j += 8) {
        const int r = blockIdx.x * 32 + ly + j;   // transposed coords
        const int c = blockIdx.y * 32 + lx;
        if (r < C && c < R) dst[(long long)r * R + c] = t[lx][ly + j];
    }
}

__global__ void k_mulvec_bf(const float* __restrict__ x, const float* __restrict__ w,
                            bf16* __restrict__ y, int n, int Dn) {
    int i = blockIdx.x * 256 + threadIdx.x;
    if (i < n) y[i] = (bf16)(x[i] * w[i % Dn]);
}

// out[row] = dot(X[row, :Dn], w)  -- one wave per row
__global__ void k_dot_rows(const float* __restrict__ X, const float* __restrict__ w,
                           float* __restrict__ out, int rows, int Dn) {
    const int row = blockIdx.x * 8 + (threadIdx.x >> 5);
    const int lane = threadIdx.x & 31;
    if (row >= rows) return;
    const float* x = X + (long long)row * Dn;
    float s = 0.f;
    for (int i = lane; i < Dn; i += 32) s += x[i] * w[i];
    for (int off = 16; off > 0; off >>= 1) s += __shfl_xor(s, off);
    if (lane == 0) out[row] = s;
}

// score[b,c,q] += s0[b,c] + s1[b,q]; masked -> += NEG
__global__ void k_score_build(float* __restrict__ S, const float* __restrict__ s0,
                              const float* __restrict__ s1,
                              const unsigned char* __restrict__ cm,
                              const unsigned char* __restrict__ qm,
                              int LCn, int LQn, int n) {
    int i = blockIdx.x * 256 + threadIdx.x;
    if (i >= n) return;
    const int q = i % LQn;
    const int t = i / LQn;
    const int c = t % LCn;
    const int b = t / LCn;
    float v = S[i] + s0[b * LCn + c] + s1[b * LQn + q];
    const bool m = (cm[b * LCn + c] != 0) && (qm[b * LQn + q] != 0);
    S[i] = m ? v : (v + NEGV);
}

// Row softmax over `cols`, bf16 output. rows = B * Hh * Lq (blockIdx.x).
__global__ __launch_bounds__(256) void k_softmax_rows(
    const float* __restrict__ S, bf16* __restrict__ O, int cols, int Lq, int Hh,
    const unsigned char* __restrict__ km, const unsigned char* __restrict__ rm)
{
    const int row = blockIdx.x;
    const int q = row % Lq;
    const int b = (row / Lq) / Hh;
    const float* src = S + (long long)row * cols;
    const bool rv = rm ? (rm[b * Lq + q] != 0) : true;
    __shared__ float red[256];
    float vl[4];
    float mx = -3.4e38f;
    int cnt = 0;
    for (int c = threadIdx.x; c < cols; c += 256, ++cnt) {
        float v = src[c];
        const bool kv = km ? (km[(long long)b * cols + c] != 0) : true;
        if (!(kv && rv)) v = NEGV;
        vl[cnt] = v;
        mx = fmaxf(mx, v);
    }
    mx = blk_max(mx, red);
    float s = 0.f;
    cnt = 0;
    for (int c = threadIdx.x; c < cols; c += 256, ++cnt) {
        float e = __expf(vl[cnt] - mx);
        vl[cnt] = e;
        s += e;
    }
    s = blk_sum(s, red);
    const float inv = 1.f / s;
    bf16* dst = O + (long long)row * cols;
    cnt = 0;
    for (int c = threadIdx.x; c < cols; c += 256, ++cnt) dst[c] = (bf16)(vl[cnt] * inv);
}

// Column softmax of S[b, :, q] over LCn; output kept [b, c, q] layout ([N,K] for
// the downstream transposed-B GEMM): O[((b*LCn + c)*LQn) + q]  (bf16)
__global__ __launch_bounds__(256) void k_softmax_colsT(
    const float* __restrict__ S, bf16* __restrict__ O, int LCn, int LQn,
    const unsigned char* __restrict__ cm)
{
    const int q = blockIdx.x;
    const int b = blockIdx.y;
    const float* src = S + (long long)b * LCn * LQn + q;
    __shared__ float red[256];
    float vl[4];
    float mx = -3.4e38f;
    int cnt = 0;
    for (int c = threadIdx.x; c < LCn; c += 256, ++cnt) {
        float v = src[(long long)c * LQn];
        if (cm && cm[b * LCn + c] == 0) v = NEGV;
        vl[cnt] = v;
        mx = fmaxf(mx, v);
    }
    mx = blk_max(mx, red);
    float s = 0.f;
    cnt = 0;
    for (int c = threadIdx.x; c < LCn; c += 256, ++cnt) {
        float e = __expf(vl[cnt] - mx);
        vl[cnt] = e;
        s += e;
    }
    s = blk_sum(s, red);
    const float inv = 1.f / s;
    bf16* dst = O + (long long)b * LCn * LQn + q;
    cnt = 0;
    for (int c = threadIdx.x; c < LCn; c += 256, ++cnt)
        dst[(long long)c * LQn] = (bf16)(vl[cnt] * inv);
}

// out = LN(x + y) * g + be ; dual fp32 + bf16 output. blockIdx.x = row.
__global__ __launch_bounds__(256) void k_add_ln(
    const float* __restrict__ x, const float* __restrict__ y,
    const float* __restrict__ g, const float* __restrict__ be,
    float* __restrict__ out, bf16* __restrict__ obf, int Dn)
{
    const int row = blockIdx.x;
    const float* xr = x + (long long)row * Dn;
    const float* yr = y + (long long)row * Dn;
    __shared__ float red[256];
    float vl[4];
    float s = 0.f;
    int cnt = 0;
    for (int i = threadIdx.x; i < Dn; i += 256, ++cnt) {
        float v = xr[i] + yr[i];
        vl[cnt] = v;
        s += v;
    }
    s = blk_sum(s, red);
    const float mean = s / (float)Dn;
    float s2 = 0.f;
    cnt = 0;
    for (int i = threadIdx.x; i < Dn; i += 256, ++cnt) {
        float d = vl[cnt] - mean;
        s2 += d * d;
    }
    s2 = blk_sum(s2, red);
    const float inv = rsqrtf(s2 / (float)Dn + 1e-6f);
    cnt = 0;
    for (int i = threadIdx.x; i < Dn; i += 256, ++cnt) {
        float o = (vl[cnt] - mean) * inv * g[i] + be[i];
        out[(long long)row * Dn + i] = o;
        obf[(long long)row * Dn + i] = (bf16)o;
    }
}

// X = concat([ctx, c2q, ctx*c2q, ctx*q2c]) as bf16; i indexes B*LC*D
__global__ void k_build_x(const float* __restrict__ ctx, const float* __restrict__ c2q,
                          const float* __restrict__ q2c, bf16* __restrict__ X,
                          int n, int Dn) {
    int i = blockIdx.x * 256 + threadIdx.x;
    if (i >= n) return;
    const int d = i % Dn;
    const long long row = i / Dn;
    const float c = ctx[i];
    const float a = c2q[i];
    const float q = q2c[i];
    bf16* xr = X + row * (long long)(4 * Dn);
    xr[d]          = (bf16)c;
    xr[Dn + d]     = (bf16)a;
    xr[2 * Dn + d] = (bf16)(c * a);
    xr[3 * Dn + d] = (bf16)(c * q);
}

// out = concat([seq(768), att(1536), ke(768)]) fp32; i indexes B*LC*3072
__global__ void k_concat_out(const float* __restrict__ seq, const float* __restrict__ att,
                             const float* __restrict__ ke, float* __restrict__ out,
                             int n, int Dn) {
    int i = blockIdx.x * 256 + threadIdx.x;
    if (i >= n) return;
    const int d = i % (4 * Dn);
    const long long row = i / (4 * Dn);
    float v;
    if (d < Dn)            v = seq[row * Dn + d];
    else if (d < 3 * Dn)   v = att[row * (2LL * Dn) + (d - Dn)];
    else                   v = ke[row * Dn + (d - 3 * Dn)];
    out[i] = v;
}

// ---------------------------------------------------------------------------
// Host orchestration
// ---------------------------------------------------------------------------
extern "C" void kernel_launch(void* const* d_in, const int* in_sizes, int n_in,
                              void* d_out, int out_size, void* d_ws, size_t ws_size,
                              hipStream_t stream) {
    (void)in_sizes; (void)n_in; (void)out_size; (void)ws_size;

    const int Bb = 8, LC = 512, LQ = 160, Dn = 768, Hh = 8, DK = 64, LK = 512, NLn = 2;
    const int HD = Hh * DK;  // 512

    const float* seq   = (const float*)d_in[0];
    const float* qry   = (const float*)d_in[1];
    const float* evd   = (const float*)d_in[2];
    const float* knw   = (const float*)d_in[3];
    const float* w4C   = (const float*)d_in[4];
    const float* w4Q   = (const float*)d_in[5];
    const float* w4mlu = (const float*)d_in[6];
    const float* cqaW  = (const float*)d_in[7];
    const float* cqab  = (const float*)d_in[8];
    // d_in[9..14] = sa_* : outer knowledge self-attn, result unused downstream -> skipped
    const float* LsWq  = (const float*)d_in[15];
    const float* LsWk  = (const float*)d_in[16];
    const float* LsWv  = (const float*)d_in[17];
    const float* LsWfc = (const float*)d_in[18];
    const float* Ln1g  = (const float*)d_in[19];
    const float* Ln1b  = (const float*)d_in[20];
    const float* LcWq  = (const float*)d_in[21];
    const float* LcWk  = (const float*)d_in[22];
    const float* LcWv  = (const float*)d_in[23];
    const float* LcWfc = (const float*)d_in[24];
    const float* Ln2g  = (const float*)d_in[25];
    const float* Ln2b  = (const float*)d_in[26];
    const unsigned char* smask = (const unsigned char*)d_in[27];
    const unsigned char* qmask = (const unsigned char*)d_in[28];
    const unsigned char* emask = (const unsigned char*)d_in[29];
    const unsigned char* kmask = (const unsigned char*)d_in[30];
    float* out = (float*)d_out;

    // ---- workspace bump allocator -------------------------------------------------
    char* ws = (char*)d_ws;
    size_t off = 0;
    auto alloc = [&](size_t bytes) -> char* {
        char* p = ws + off;
        off += (bytes + 255) & ~(size_t)255;
        return p;
    };

    // persistent buffers
    bf16*  qry_bf = (bf16*)alloc((size_t)Bb * LQ * Dn * 2);
    bf16*  evd_bf = (bf16*)alloc((size_t)Bb * LQ * Dn * 2);
    float* att    = (float*)alloc((size_t)Bb * LC * 2 * Dn * 4);
    bf16*  att_bf = (bf16*)alloc((size_t)Bb * LC * 2 * Dn * 2);
    float* keb    = (float*)alloc((size_t)Bb * LK * Dn * 4);
    bf16*  ke_bf  = (bf16*)alloc((size_t)Bb * LK * Dn * 2);
    float* sob    = (float*)alloc((size_t)Bb * LK * Dn * 4);
    bf16*  so_bf  = (bf16*)alloc((size_t)Bb * LK * Dn * 2);
    bf16*  wt_bf  = (bf16*)alloc((size_t)Dn * 4 * Dn * 2);   // max weight: 768x3072
    float* s0     = (float*)alloc((size_t)Bb * LC * 4);
    float* s1     = (float*)alloc((size_t)Bb * LQ * 4);
    const size_t tmp_mark = off;

    // phase-1 (_s2q) temporaries
    bf16*  ctxw_bf = (bf16*)alloc((size_t)Bb * LC * Dn * 2);
    bf16*  ctxT_bf = (bf16*)alloc((size_t)Bb * Dn * LC * 2);   // [B, D, LC]
    bf16*  qinT_bf = (bf16*)alloc((size_t)Bb * Dn * LQ * 2);   // [B, D, LQ]
    float* score   = (float*)alloc((size_t)Bb * LC * LQ * 4);
    bf16*  scq_bf  = (bf16*)alloc((size_t)Bb * LC * LQ * 2);
    bf16*  sctT_bf = (bf16*)alloc((size_t)Bb * LC * LQ * 2);   // [B, LC, LQ]
    float* c2q     = (float*)alloc((size_t)Bb * LC * Dn * 4);
    float* q2c     = (float*)alloc((size_t)Bb * LC * Dn * 4);
    bf16*  tmp_bf  = (bf16*)alloc((size_t)Bb * LC * LC * 2);
    bf16*  X_bf    = (bf16*)alloc((size_t)Bb * LC * 4 * Dn * 2);

    auto gemm = [&](bool outBF, const bf16* A, const bf16* Bm, void* C,
                    int M, int N, int K, int lda, int ldb, int ldc,
                    long long sAb, long long sAh, long long sBb, long long sBh,
                    long long sCb, long long sCh, int nb, int H2, float alpha,
                    const float* bias) {
        dim3 g((N + 63) / 64, (M + 127) / 128, nb * H2);
        if (outBF)
            gemm_wmma_bf16<true><<<g, 256, 0, stream>>>(A, Bm, C, M, N, K, lda, ldb,
                ldc, sAb, sAh, sBb, sBh, sCb, sCh, H2, alpha, bias);
        else
            gemm_wmma_bf16<false><<<g, 256, 0, stream>>>(A, Bm, C, M, N, K, lda, ldb,
                ldc, sAb, sAh, sBb, sBh, sCb, sCh, H2, alpha, bias);
    };
    auto cvt = [&](const float* x, bf16* y, int n) {
        k_f2bf<<<(n + 255) / 256, 256, 0, stream>>>(x, y, n);
    };
    auto cvtT = [&](const float* x, bf16* y, int Bt, int R, int C) {
        int n = Bt * R * C;
        k_cvtT<<<(n + 255) / 256, 256, 0, stream>>>(x, y, Bt, R, C);
    };

    // ---- phase 0: conversions -----------------------------------------------------
    cvt(qry, qry_bf, Bb * LQ * Dn);
    cvt(evd, evd_bf, Bb * LQ * Dn);
    cvtT(seq, ctxT_bf, Bb, LC, Dn);                       // ctx^T for q2c
    k_mulvec_bf<<<(Bb * LC * Dn + 255) / 256, 256, 0, stream>>>(seq, w4mlu, ctxw_bf,
                                                                Bb * LC * Dn, Dn);
    k_dot_rows<<<(Bb * LC + 7) / 8, 256, 0, stream>>>(seq, w4C, s0, Bb * LC, Dn);
    cvt(cqaW, wt_bf, Dn * 4 * Dn);   // cqa_W [768,3072] already [N,K] for X @ W^T

    // ---- phase 1: _s2q for query and evidence -------------------------------------
    auto s2q = [&](const float* qin, const bf16* qin_bf, const unsigned char* qm,
                   float* att_col) {
        k_dot_rows<<<(Bb * LQ + 7) / 8, 256, 0, stream>>>(qin, w4Q, s1, Bb * LQ, Dn);
        cvtT(qin, qinT_bf, Bb, LQ, Dn);                   // qin^T for c2q
        // s2 = (ctx*w4mlu) @ qin^T   (B = qin [LQ, D] as [N,K])
        gemm(false, ctxw_bf, qin_bf, score, LC, LQ, Dn, Dn, Dn, LQ,
             (long long)LC * Dn, 0, (long long)LQ * Dn, 0, (long long)LC * LQ, 0,
             Bb, 1, 1.f, nullptr);
        const int n = Bb * LC * LQ;
        k_score_build<<<(n + 255) / 256, 256, 0, stream>>>(score, s0, s1, smask, qm,
                                                           LC, LQ, n);
        k_softmax_rows<<<Bb * LC, 256, 0, stream>>>(score, scq_bf, LQ, LC, 1, qm, smask);
        dim3 gt(LQ, Bb);
        k_softmax_colsT<<<gt, 256, 0, stream>>>(score, sctT_bf, LC, LQ, smask);
        // c2q = score_c @ qin         (B = qin^T [D, LQ] as [N,K])
        gemm(false, scq_bf, qinT_bf, c2q, LC, Dn, LQ, LQ, LQ, Dn,
             (long long)LC * LQ, 0, (long long)Dn * LQ, 0, (long long)LC * Dn, 0,
             Bb, 1, 1.f, nullptr);
        // tmp = score_c @ score_t     (B = score_t^T = sctT [LC, LQ] as [N,K])
        gemm(true, scq_bf, sctT_bf, tmp_bf, LC, LC, LQ, LQ, LQ, LC,
             (long long)LC * LQ, 0, (long long)LC * LQ, 0, (long long)LC * LC, 0,
             Bb, 1, 1.f, nullptr);
        // q2c = tmp @ ctx             (B = ctx^T [D, LC] as [N,K])
        gemm(false, tmp_bf, ctxT_bf, q2c, LC, Dn, LC, LC, LC, Dn,
             (long long)LC * LC, 0, (long long)Dn * LC, 0, (long long)LC * Dn, 0,
             Bb, 1, 1.f, nullptr);
        const int n2 = Bb * LC * Dn;
        k_build_x<<<(n2 + 255) / 256, 256, 0, stream>>>(seq, c2q, q2c, X_bf, n2, Dn);
        // att_col = X @ cqa_W^T + cqa_b   (strided into att, ldc = 1536)
        gemm(false, X_bf, wt_bf, att_col, LC, Dn, 4 * Dn, 4 * Dn, 4 * Dn, 2 * Dn,
             (long long)LC * 4 * Dn, 0, 0, 0, (long long)LC * 2 * Dn, 0,
             Bb, 1, 1.f, cqab);
    };
    s2q(qry, qry_bf, qmask, att);
    s2q(evd, evd_bf, emask, att + Dn);
    cvt(att, att_bf, Bb * LC * 2 * Dn);

    // ---- phase 2 temporaries overlay phase-1 temporaries --------------------------
    off = tmp_mark;
    bf16*  qb      = (bf16*)alloc((size_t)Bb * LK * HD * 2);
    bf16*  kb      = (bf16*)alloc((size_t)Bb * LK * HD * 2);
    bf16*  vb      = (bf16*)alloc((size_t)Bb * LK * HD * 2);
    bf16*  vbT     = (bf16*)alloc((size_t)HD * Bb * LK * 2);  // [HD, B*keys]
    float* scores  = (float*)alloc((size_t)Bb * Hh * LK * LK * 4);
    bf16*  attn_bf = (bf16*)alloc((size_t)Bb * Hh * LK * LK * 2);
    bf16*  ctxo_bf = (bf16*)alloc((size_t)Bb * LK * HD * 2);
    float* fco     = (float*)alloc((size_t)Bb * LK * Dn * 4);

    // ---- phase 2: knowledge attention stack ---------------------------------------
    const float* ke_f = knw;
    cvt(knw, ke_bf, Bb * LK * Dn);
    const float scal = 0.125f;  // log_512(512) / sqrt(64)
    const long long BKEY = (long long)Bb * LK;   // 4096 (keys total, self & cross)
    dim3 trg((HD + 31) / 32, (Bb * LK + 31) / 32);

    for (int i = 0; i < NLn; ++i) {
        // ---- self-attention on ke ----
        cvtT(LsWq + (size_t)i * Dn * HD, wt_bf, 1, Dn, HD);  // -> [HD, Dn]
        gemm(true, ke_bf, wt_bf, qb, Bb * LK, HD, Dn, Dn, Dn, HD,
             0, 0, 0, 0, 0, 0, 1, 1, 1.f, nullptr);
        cvtT(LsWk + (size_t)i * Dn * HD, wt_bf, 1, Dn, HD);
        gemm(true, ke_bf, wt_bf, kb, Bb * LK, HD, Dn, Dn, Dn, HD,
             0, 0, 0, 0, 0, 0, 1, 1, 1.f, nullptr);
        cvtT(LsWv + (size_t)i * Dn * HD, wt_bf, 1, Dn, HD);
        gemm(true, ke_bf, wt_bf, vb, Bb * LK, HD, Dn, Dn, Dn, HD,
             0, 0, 0, 0, 0, 0, 1, 1, 1.f, nullptr);
        k_tr_bf<<<trg, 256, 0, stream>>>(vb, vbT, Bb * LK, HD);
        // scores[b,h] = scal * q[b,:,h] @ k[b,:,h]^T
        gemm(false, qb, kb, scores, LK, LK, DK, HD, HD, LK,
             (long long)LK * HD, 64, (long long)LK * HD, 64,
             (long long)Hh * LK * LK, (long long)LK * LK, Bb, Hh, scal, nullptr);
        k_softmax_rows<<<Bb * Hh * LK, 256, 0, stream>>>(scores, attn_bf, LK, LK, Hh,
                                                         kmask, kmask);
        // ctx = attn @ v              (B = v^T slice [64, keys] as [N,K])
        gemm(true, attn_bf, vbT, ctxo_bf, LK, DK, LK, LK, (int)BKEY, HD,
             (long long)Hh * LK * LK, (long long)LK * LK,
             LK, 64LL * BKEY,
             (long long)LK * HD, 64, Bb, Hh, 1.f, nullptr);
        cvtT(LsWfc + (size_t)i * HD * Dn, wt_bf, 1, HD, Dn);  // -> [Dn, HD]
        gemm(false, ctxo_bf, wt_bf, fco, Bb * LK, Dn, HD, HD, HD, Dn,
             0, 0, 0, 0, 0, 0, 1, 1, 1.f, nullptr);
        k_add_ln<<<Bb * LK, 256, 0, stream>>>(fco, ke_f, Ln1g + (size_t)i * Dn,
                                              Ln1b + (size_t)i * Dn, sob, so_bf, Dn);

        // ---- cross-attention: q from so, k/v from att ----
        cvtT(LcWq + (size_t)i * Dn * HD, wt_bf, 1, Dn, HD);
        gemm(true, so_bf, wt_bf, qb, Bb * LK, HD, Dn, Dn, Dn, HD,
             0, 0, 0, 0, 0, 0, 1, 1, 1.f, nullptr);
        cvtT(LcWk + (size_t)i * 2 * Dn * HD, wt_bf, 1, 2 * Dn, HD);
        gemm(true, att_bf, wt_bf, kb, Bb * LC, HD, 2 * Dn, 2 * Dn, 2 * Dn, HD,
             0, 0, 0, 0, 0, 0, 1, 1, 1.f, nullptr);
        cvtT(LcWv + (size_t)i * 2 * Dn * HD, wt_bf, 1, 2 * Dn, HD);
        gemm(true, att_bf, wt_bf, vb, Bb * LC, HD, 2 * Dn, 2 * Dn, 2 * Dn, HD,
             0, 0, 0, 0, 0, 0, 1, 1, 1.f, nullptr);
        k_tr_bf<<<trg, 256, 0, stream>>>(vb, vbT, Bb * LC, HD);
        gemm(false, qb, kb, scores, LK, LC, DK, HD, HD, LC,
             (long long)LK * HD, 64, (long long)LC * HD, 64,
             (long long)Hh * LK * LC, (long long)LK * LC, Bb, Hh, scal, nullptr);
        k_softmax_rows<<<Bb * Hh * LK, 256, 0, stream>>>(scores, attn_bf, LC, LK, Hh,
                                                         smask, smask);
        gemm(true, attn_bf, vbT, ctxo_bf, LK, DK, LC, LC, (int)BKEY, HD,
             (long long)Hh * LK * LC, (long long)LK * LC,
             LC, 64LL * BKEY,
             (long long)LK * HD, 64, Bb, Hh, 1.f, nullptr);
        cvtT(LcWfc + (size_t)i * HD * Dn, wt_bf, 1, HD, Dn);
        gemm(false, ctxo_bf, wt_bf, fco, Bb * LK, Dn, HD, HD, HD, Dn,
             0, 0, 0, 0, 0, 0, 1, 1, 1.f, nullptr);
        k_add_ln<<<Bb * LK, 256, 0, stream>>>(fco, sob, Ln2g + (size_t)i * Dn,
                                              Ln2b + (size_t)i * Dn, keb, ke_bf, Dn);
        ke_f = keb;
    }

    // ---- final concat -------------------------------------------------------------
    const int n3 = Bb * LC * 4 * Dn;
    k_concat_out<<<(n3 + 255) / 256, 256, 0, stream>>>(seq, att, keb, out, n3, Dn);
}
